// SingleHeadSelfAttention_80058190397974
// MI455X (gfx1250) — matrix operand order repaired and checked by reading, not compiled, not run on hardware
//
#include <hip/hip_runtime.h>
#include <hip/hip_bf16.h>
#include <math.h>

#define TB 8
#define TS 2048
#define TD 1024
#define NEGVAL -999999.0f

typedef __bf16 bf16x16 __attribute__((ext_vector_type(16)));
typedef float  f32x8   __attribute__((ext_vector_type(8)));
typedef unsigned int   uint4v __attribute__((ext_vector_type(4)));
typedef float          f32x4  __attribute__((ext_vector_type(4)));
typedef unsigned short u16x4  __attribute__((ext_vector_type(4)));
typedef unsigned int   u32x4  __attribute__((ext_vector_type(4)));
typedef int            i32x8  __attribute__((ext_vector_type(8)));
typedef int            i32x4  __attribute__((ext_vector_type(4)));

union Frag { uint4v u[2]; bf16x16 b; };

__device__ inline unsigned short f32_to_bf16_bits(float f) {
  union { float f; unsigned u; } x; x.f = f;
  unsigned r = x.u + 0x7FFFu + ((x.u >> 16) & 1u);  // round-to-nearest-even
  return (unsigned short)(r >> 16);
}

// A-fragment: 16x32 bf16 tile stored row-major [rows][32] in LDS.
// lane row = row0 + lane%16 ; grp = lane/16 selects K {0-7,16-23} vs {8-15,24-31}
__device__ inline bf16x16 load_fragA(const uint4v* t, int row0, int lane) {
  int m = row0 + (lane & 15);
  int g = (lane >> 4) & 1;
  Frag f;
  f.u[0] = t[m * 4 + g];
  f.u[1] = t[m * 4 + 2 + g];
  return f.b;
}

// B-fragment: B is [32 k][16 n]; staged transposed as [n][32 k] in LDS.
// lane col = row0 + lane%16 ; grp = lane/16 selects K {0-7,16-23} vs {8-15,24-31}
__device__ inline bf16x16 load_fragB(const uint4v* t, int row0, int lane) {
  int n = row0 + (lane & 15);
  int g = (lane >> 4) & 1;
  Frag f;
  f.u[0] = t[n * 4 + g];
  f.u[1] = t[n * 4 + 2 + g];
  return f.b;
}

__device__ inline f32x8 wmma_bf16(bf16x16 a, bf16x16 b, f32x8 c) {
  return __builtin_amdgcn_wmma_f32_16x16x32_bf16(false, a, false, b,
                                                 (short)0, c, false, false);
}

// ---------------------------------------------------------------------------
// Tensor Data Mover: DMA a 2D bf16 tile (tile_w x tile_h elements) from a
// row-major global tensor (row pitch = stride_elems) into LDS at l_dst.
// D# packing per CDNA5 ISA 8.3/8.4 (group0: count|lds|global|type=2,
// group1: data_size=2B, dims/strides; groups 2-3 zero for 2D tiles).
// 6-arg builtin form on this toolchain: extra int32x8 before cpol.
// Issue from ONE wave; completion via s_wait_tensorcnt.
// ---------------------------------------------------------------------------
__device__ inline void tdm_load_tile_bf16(const unsigned short* g_tile,
                                          unsigned short* l_dst,
                                          unsigned stride_elems,
                                          unsigned tile_w, unsigned tile_h,
                                          unsigned rows_total) {
  unsigned long long ga = (unsigned long long)(uintptr_t)g_tile;  // byte VA
  unsigned lds_off = (unsigned)(uintptr_t)l_dst;  // generic low 32 = LDS offs
  u32x4 g0;
  g0.x = 1u;                                             // count = 1
  g0.y = lds_off;                                        // lds_addr
  g0.z = (unsigned)ga;                                   // global_addr[31:0]
  g0.w = (unsigned)((ga >> 32) & 0x1FFFFFFull)           // global_addr[56:32]
         | 0x80000000u;                                  // type = 2 ("image")
  i32x8 g1;
  g1[0] = 0x00010000;                                    // data_size=1 (2B)
  g1[1] = (int)(stride_elems << 16);                     // tensor_dim0[15:0]
  g1[2] = (int)((stride_elems >> 16) | (rows_total << 16)); // d0 hi | dim1 lo
  g1[3] = (int)((rows_total >> 16) | (tile_w << 16));    // dim1 hi | tile_dim0
  g1[4] = (int)tile_h;                                   // tile_dim1 (dim2=0)
  g1[5] = (int)stride_elems;                             // dim0_stride[31:0]
  g1[6] = 0;
  g1[7] = 0;
  i32x4 z4; z4[0] = 0; z4[1] = 0; z4[2] = 0; z4[3] = 0;  // groups 2/3 unused
  i32x8 z8;
#pragma unroll
  for (int e = 0; e < 8; ++e) z8[e] = 0;
  __builtin_amdgcn_tensor_load_to_lds(g0, g1, z4, z4, z8, 0);
}

// ---------------------------------------------------------------------------
// Kernel 1: Out(bf16)[M,N] = (X(f32)[M,K] @ W(f32)[K,N] + bias) * scale
// 64x64 tile per block, 128 threads (4 waves). f32->bf16 conversion while
// staging (so no TDM here); b128 global loads + prefetch of next K-chunk.
// ---------------------------------------------------------------------------
__global__ __launch_bounds__(128)
void qkv_proj_kernel(const float* __restrict__ X, const float* __restrict__ W,
                     const float* __restrict__ bias,
                     unsigned short* __restrict__ Out,
                     float scale, int M, int K, int N) {
  __shared__ __align__(16) unsigned short Xs[64 * 32];
  __shared__ __align__(16) unsigned short WsT[64 * 32];
  const int t = threadIdx.x;
  const int lane = t & 31, wid = t >> 5;
  const int n0 = blockIdx.x * 64;
  const int m0 = blockIdx.y * 64;

  f32x8 acc[4] = {};
  for (int k0 = 0; k0 < K; k0 += 32) {
    // stage X tile (vectorized f32x4 loads, convert to bf16)
    for (int j = t; j < 512; j += 128) {
      int r = j >> 3, c = (j & 7) * 4;
      f32x4 xv = *(const f32x4*)(X + (size_t)(m0 + r) * K + k0 + c);
#pragma unroll
      for (int e = 0; e < 4; ++e) Xs[r * 32 + c + e] = f32_to_bf16_bits(xv[e]);
    }
    // stage W tile transposed
    for (int j = t; j < 512; j += 128) {
      int kk = j >> 4, nn = (j & 15) * 4;
      f32x4 wv = *(const f32x4*)(W + (size_t)(k0 + kk) * N + n0 + nn);
#pragma unroll
      for (int e = 0; e < 4; ++e)
        WsT[(nn + e) * 32 + kk] = f32_to_bf16_bits(wv[e]);
    }
    if (k0 + 32 < K) {  // prefetch next chunk -> global_prefetch_b8
      __builtin_prefetch(X + (size_t)(m0 + (t & 63)) * K + k0 + 32, 0, 1);
      __builtin_prefetch(W + (size_t)(k0 + 32 + (t & 31)) * N + n0, 0, 1);
    }
    __syncthreads();
    bf16x16 a = load_fragA((const uint4v*)Xs, wid * 16, lane);
#pragma unroll
    for (int s = 0; s < 4; ++s) {
      bf16x16 b = load_fragB((const uint4v*)WsT, s * 16, lane);
      acc[s] = wmma_bf16(a, b, acc[s]);
    }
    __syncthreads();
  }
  const int g = lane >> 4, col = lane & 15;
#pragma unroll
  for (int s = 0; s < 4; ++s) {
    int n = n0 + s * 16 + col;
    float bv = bias[n];
#pragma unroll
    for (int r = 0; r < 8; ++r) {
      int m = m0 + wid * 16 + r + 8 * g;
      Out[(size_t)m * N + n] = f32_to_bf16_bits((acc[s][r] + bv) * scale);
    }
  }
}

// ---------------------------------------------------------------------------
// Kernel 2: causal softmax(Q Kᵀ) -> P(bf16)[B,S,S].
// Q/K tiles DMA'd by the Tensor Data Mover into double-buffered LDS; wave 0
// issues the next chunk's DMA while all waves run WMMAs on the current one.
// Pass 0: streaming row max / sum-exp. Pass 1: recompute, write P.
// ---------------------------------------------------------------------------
__global__ __launch_bounds__(128)
void attn_softmax_kernel(const unsigned short* __restrict__ Q,
                         const unsigned short* __restrict__ Km,
                         unsigned short* __restrict__ P) {
  __shared__ __align__(16) unsigned short Qs[2][64 * 32];
  __shared__ __align__(16) unsigned short Ks[2][64 * 32];
  const int t = threadIdx.x;
  const int lane = t & 31, wid = t >> 5;
  const int qt = blockIdx.x;
  const int b  = blockIdx.y;
  const int q0 = qt * 64;
  const unsigned short* Qb = Q + (size_t)b * TS * TD;
  const unsigned short* Kb = Km + (size_t)b * TS * TD;
  unsigned short* Pb = P + (size_t)b * TS * TS;

  const int g = lane >> 4, col = lane & 15;
  float mrow[8], lrow[8];
#pragma unroll
  for (int r = 0; r < 8; ++r) { mrow[r] = -3.0e38f; lrow[r] = 0.0f; }

  const int nch = TD / 32;
  for (int pass = 0; pass < 2; ++pass) {
    for (int kt = 0; kt <= qt; ++kt) {                 // causal: skip kt > qt
      const int k0 = kt * 64;
      f32x8 acc[4] = {};
      if (wid == 0) {                                   // prologue DMA chunk 0
        tdm_load_tile_bf16(Qb + (size_t)q0 * TD, Qs[0], TD, 32, 64, TS);
        tdm_load_tile_bf16(Kb + (size_t)k0 * TD, Ks[0], TD, 32, 64, TS);
      }
      for (int i = 0; i < nch; ++i) {
        const int cur = i & 1;
        if (wid == 0) __builtin_amdgcn_s_wait_tensorcnt(0);
        __syncthreads();   // cur buffer ready; other buffer's reads drained
        if (wid == 0 && i + 1 < nch) {                  // DMA next chunk
          const int d1 = (i + 1) * 32;
          tdm_load_tile_bf16(Qb + (size_t)q0 * TD + d1, Qs[cur ^ 1],
                             TD, 32, 64, TS);
          tdm_load_tile_bf16(Kb + (size_t)k0 * TD + d1, Ks[cur ^ 1],
                             TD, 32, 64, TS);
        }
        bf16x16 a = load_fragA((const uint4v*)Qs[cur], wid * 16, lane);
#pragma unroll
        for (int s = 0; s < 4; ++s) {
          bf16x16 kb = load_fragB((const uint4v*)Ks[cur], s * 16, lane);
          acc[s] = wmma_bf16(a, kb, acc[s]);
        }
      }
      __syncthreads();      // drain last frag reads before next kt's DMA
      if (kt == qt) {                                   // mask diagonal tile
#pragma unroll
        for (int s = 0; s < 4; ++s) {
          int kg = k0 + s * 16 + col;
#pragma unroll
          for (int r = 0; r < 8; ++r) {
            int qg = q0 + wid * 16 + r + 8 * g;
            if (kg > qg) acc[s][r] = NEGVAL;
          }
        }
      }
      if (pass == 0) {                                  // streaming stats
#pragma unroll
        for (int r = 0; r < 8; ++r) {
          float v0 = fmaxf(fmaxf(acc[0][r], acc[1][r]),
                           fmaxf(acc[2][r], acc[3][r]));
#pragma unroll
          for (int off = 1; off < 16; off <<= 1)
            v0 = fmaxf(v0, __shfl_xor(v0, off, 16));
          float mnew = fmaxf(mrow[r], v0);
          float se = 0.0f;
#pragma unroll
          for (int s = 0; s < 4; ++s) se += __expf(acc[s][r] - mnew);
#pragma unroll
          for (int off = 1; off < 16; off <<= 1)
            se += __shfl_xor(se, off, 16);
          lrow[r] = lrow[r] * __expf(mrow[r] - mnew) + se;
          mrow[r] = mnew;
        }
      } else {                                          // write normalized P
#pragma unroll
        for (int s = 0; s < 4; ++s) {
          int kg = k0 + s * 16 + col;
#pragma unroll
          for (int r = 0; r < 8; ++r) {
            int qg = q0 + wid * 16 + r + 8 * g;
            float p = __expf(acc[s][r] - mrow[r]) / lrow[r];
            Pb[(size_t)qg * TS + kg] = f32_to_bf16_bits(p);
          }
        }
      }
    }
  }
}

// ---------------------------------------------------------------------------
// Kernel 3: Out(f32)[B,S,D] = P(bf16) @ V(bf16), causal-skipped.
// P tiles via TDM (double-buffered); V tiles manually transpose-staged into
// the same double-buffer schedule (b64 global loads).
// ---------------------------------------------------------------------------
__global__ __launch_bounds__(128)
void pv_kernel(const unsigned short* __restrict__ P,
               const unsigned short* __restrict__ V,
               float* __restrict__ Out) {
  __shared__ __align__(16) unsigned short Ps[2][64 * 32];
  __shared__ __align__(16) unsigned short VsT[2][64 * 32];
  const int t = threadIdx.x;
  const int lane = t & 31, wid = t >> 5;
  const int d0 = blockIdx.x * 64;
  const int qt = blockIdx.y;
  const int b  = blockIdx.z;
  const int q0 = qt * 64;
  const unsigned short* Pb = P + (size_t)b * TS * TS;
  const unsigned short* Vb = V + (size_t)b * TS * TD;
  float* Ob = Out + (size_t)b * TS * TD;

  f32x8 acc[4] = {};
  const int nch = ((qt + 1) * 64) / 32;                 // P zero past diagonal
  // prologue: chunk 0
  if (wid == 0)
    tdm_load_tile_bf16(Pb + (size_t)q0 * TS, Ps[0], TS, 32, 64, TS);
  for (int j = t; j < 512; j += 128) {                  // stage V chunk 0 (T)
    int kk = j >> 4, nn = (j & 15) * 4;
    u16x4 vv = *(const u16x4*)(Vb + (size_t)kk * TD + d0 + nn);
#pragma unroll
    for (int e = 0; e < 4; ++e) VsT[0][(nn + e) * 32 + kk] = vv[e];
  }
  for (int i = 0; i < nch; ++i) {
    const int cur = i & 1;
    if (wid == 0) __builtin_amdgcn_s_wait_tensorcnt(0);
    __syncthreads();
    if (i + 1 < nch) {                                  // pipeline next chunk
      const int k1 = (i + 1) * 32;
      if (wid == 0)
        tdm_load_tile_bf16(Pb + (size_t)q0 * TS + k1, Ps[cur ^ 1],
                           TS, 32, 64, TS);
      for (int j = t; j < 512; j += 128) {
        int kk = j >> 4, nn = (j & 15) * 4;
        u16x4 vv = *(const u16x4*)(Vb + (size_t)(k1 + kk) * TD + d0 + nn);
#pragma unroll
        for (int e = 0; e < 4; ++e) VsT[cur ^ 1][(nn + e) * 32 + kk] = vv[e];
      }
    }
    bf16x16 a = load_fragA((const uint4v*)Ps[cur], wid * 16, lane);
#pragma unroll
    for (int s = 0; s < 4; ++s) {
      bf16x16 bb = load_fragB((const uint4v*)VsT[cur], s * 16, lane);
      acc[s] = wmma_bf16(a, bb, acc[s]);
    }
  }
  const int g = lane >> 4, col = lane & 15;
#pragma unroll
  for (int s = 0; s < 4; ++s) {
    int n = d0 + s * 16 + col;
#pragma unroll
    for (int r = 0; r < 8; ++r) {
      int m = q0 + wid * 16 + r + 8 * g;
      Ob[(size_t)m * TD + n] = acc[s][r];
    }
  }
}

// ---------------------------------------------------------------------------
extern "C" void kernel_launch(void* const* d_in, const int* in_sizes, int n_in,
                              void* d_out, int out_size, void* d_ws,
                              size_t ws_size, hipStream_t stream) {
  (void)in_sizes; (void)n_in; (void)out_size; (void)ws_size;
  const float* x  = (const float*)d_in[0];
  // d_in[1] is the causal mask; causality is applied analytically instead.
  const float* Wq = (const float*)d_in[2];
  const float* bq = (const float*)d_in[3];
  const float* Wk = (const float*)d_in[4];
  const float* bk = (const float*)d_in[5];
  const float* Wv = (const float*)d_in[6];
  const float* bv = (const float*)d_in[7];
  float* out = (float*)d_out;

  const size_t qkv_elems = (size_t)TB * TS * TD;        // 16M bf16 each
  unsigned short* q = (unsigned short*)d_ws;
  unsigned short* k = q + qkv_elems;
  unsigned short* v = k + qkv_elems;
  unsigned short* P = v + qkv_elems;                    // B*S*S bf16 = 64 MB

  dim3 blk(128);
  dim3 gproj(TD / 64, (TB * TS) / 64);
  const float scale = 1.0f / sqrtf((float)TD);          // fold 1/sqrt(D) into Q
  qkv_proj_kernel<<<gproj, blk, 0, stream>>>(x, Wq, bq, q, scale, TB * TS, TD, TD);
  qkv_proj_kernel<<<gproj, blk, 0, stream>>>(x, Wk, bk, k, 1.0f,  TB * TS, TD, TD);
  qkv_proj_kernel<<<gproj, blk, 0, stream>>>(x, Wv, bv, v, 1.0f,  TB * TS, TD, TD);

  dim3 gattn(TS / 64, TB);
  attn_softmax_kernel<<<gattn, blk, 0, stream>>>(q, k, P);

  dim3 gpv(TD / 64, TS / 64, TB);
  pv_kernel<<<gpv, blk, 0, stream>>>(P, v, out);
}
